// DIMKT_CC_12352325943861
// MI455X (gfx1250) — matrix operand order repaired and checked by reading, not compiled, not run on hardware
//
#include <hip/hip_runtime.h>
#include <hip/hip_bf16.h>

// ---------------------------------------------------------------------------
// DIMKT on MI455X (gfx1250): bf16 WMMA GEMMs, LDS-resident recurrent state,
// register-resident weights in the sequential scan (latency-bound phase).
// ---------------------------------------------------------------------------

#define B_DIM 512
#define S_DIM 200
#define E_DIM 128

typedef __bf16 bf16;
typedef __attribute__((ext_vector_type(8)))  __bf16 v8bf;
typedef __attribute__((ext_vector_type(16))) __bf16 v16bf;
typedef __attribute__((ext_vector_type(8)))  float  v8f;

// --- WMMA fragment helpers --------------------------------------------------
// A fragment: 16x32 bf16 tile, rows in LDS (row-major, `stride` bf16 per row).
// ISA layout: lanes 0-15 -> M=lane, K in {k0..k0+7, k0+16..k0+23};
//             lanes 16-31 -> M=lane-16, K in {k0+8..k0+15, k0+24..k0+31}.
__device__ __forceinline__ v16bf load_a_frag(const bf16* A, int stride, int k0) {
    int lane = threadIdx.x & 31;
    int m    = lane & 15;
    int koff = k0 + ((lane >> 4) << 3);
    const bf16* p = A + m * stride + koff;
    v8bf lo = *(const v8bf*)p;           // K = koff .. koff+7
    v8bf hi = *(const v8bf*)(p + 16);    // K = koff+16 .. koff+23
    return __builtin_shufflevector(lo, hi, 0,1,2,3,4,5,6,7,8,9,10,11,12,13,14,15);
}

// B fragment: 32x16 bf16 tile from transposed weights Wt[k][128] (global/L2).
// lane -> N = n0+lane%16, the 16 vector elements -> K in the same order as the A fragment.
__device__ __forceinline__ v16bf load_b_frag(const bf16* Wt, int k0, int n0) {
    int lane = threadIdx.x & 31;
    int n    = n0 + (lane & 15);
    int koff = k0 + ((lane >> 4) << 3);
    v16bf b;
    #pragma unroll
    for (int s = 0; s < 8; s++) {
        b[s]     = Wt[(size_t)(koff + s) * E_DIM + n];
        b[s + 8] = Wt[(size_t)(koff + 16 + s) * E_DIM + n];
    }
    return b;
}

__device__ __forceinline__ v8f wmma_bf16(v16bf a, v16bf b, v8f c) {
    return __builtin_amdgcn_wmma_f32_16x16x32_bf16(
        /*neg_a=*/false, a, /*neg_b=*/false, b,
        /*c_mod=*/(short)0, c, /*reuse_a=*/false, /*reuse_b=*/false);
}

__device__ __forceinline__ float sigmoidf_fast(float x) {
    return 1.0f / (1.0f + __expf(-x));
}

// --- Prep: f32 -> bf16 conversions -----------------------------------------
__global__ void cvt_table_kernel(const float* __restrict__ src,
                                 bf16* __restrict__ dst, int n) {
    int i = blockIdx.x * 256 + threadIdx.x;
    if (i < n) dst[i] = (bf16)src[i];
}

// W: [128][kdim] row-major (out x in)  ->  Wt: [kdim][128] bf16
__global__ void cvt_weightT_kernel(const float* __restrict__ W,
                                   bf16* __restrict__ Wt, int kdim) {
    int i = blockIdx.x * 256 + threadIdx.x;
    if (i < kdim * E_DIM) {
        int k = i >> 7;
        int n = i & 127;
        Wt[i] = (bf16)W[n * kdim + k];
    }
}

// --- Phase B: input_data / target_data = gather(cat) @ W1^T + b1 ------------
// out layout: f32 [S][B][E] (step-major so the scan reads contiguously).
__global__ __launch_bounds__(256)
void w1_gemm_kernel(const int* __restrict__ cI, const int* __restrict__ sdI,
                    const bf16* __restrict__ ctb, const bf16* __restrict__ sdtb,
                    const bf16* __restrict__ W1t, const float* __restrict__ b1,
                    float* __restrict__ outbuf) {
    __shared__ bf16 Abuf[16 * 256];
    const int t  = blockIdx.y;
    const int b0 = blockIdx.x * 16;
    // Gather 16 rows of concat(c_emb, sd_emb) into LDS (bf16).
    for (int idx = threadIdx.x; idx < 16 * E_DIM; idx += 256) {
        int r = idx >> 7, e = idx & 127;
        int b   = b0 + r;
        int cix = cI[b * S_DIM + t];
        int six = sdI[b * S_DIM + t];
        Abuf[r * 256 + e]         = ctb[cix * E_DIM + e];
        Abuf[r * 256 + 128 + e]   = sdtb[six * E_DIM + e];
    }
    __syncthreads();

    const int wave = threadIdx.x >> 5;
    const int lane = threadIdx.x & 31;
    const int n0   = wave * 16;
    v8f acc = {};
    #pragma unroll
    for (int k0 = 0; k0 < 256; k0 += 32) {
        v16bf a = load_a_frag(Abuf, 256, k0);
        v16bf b = load_b_frag(W1t, k0, n0);
        acc = wmma_bf16(a, b, acc);
    }
    const int colL  = lane & 15;
    const int rbase = (lane >> 4) << 3;
    const float bias = b1[n0 + colL];
    #pragma unroll
    for (int v = 0; v < 8; v++) {
        int row = rbase + v;
        outbuf[((size_t)t * B_DIM + b0 + row) * E_DIM + n0 + colL] = acc[v] + bias;
    }
}

// --- Phase C: sequential scan (the recurrence) ------------------------------
// One WG owns 16 rows of the state k [16x128 f32 in LDS] for all 200 steps.
// 8 waves each own a 16-column N-tile; each wave keeps ALL of its weight
// B-fragments (W2..W6 -> 36 frags = 288 VGPRs) in registers across the scan,
// so the steady-state inner loop is LDS A-frag loads + v_wmma only.
__global__ __launch_bounds__(256, 1)
void dimkt_scan_kernel(const int* __restrict__ sdI, const int* __restrict__ aI,
                       const float* __restrict__ in_data,
                       const float* __restrict__ tg_data,
                       const bf16* __restrict__ atb, const bf16* __restrict__ sdtb,
                       const bf16* __restrict__ W2t, const bf16* __restrict__ W3t,
                       const bf16* __restrict__ W4t, const bf16* __restrict__ W5t,
                       const bf16* __restrict__ W6t,
                       const float* __restrict__ b2, const float* __restrict__ b3,
                       const float* __restrict__ b4, const float* __restrict__ b5,
                       const float* __restrict__ b6,
                       const float* __restrict__ knowledge,
                       float* __restrict__ out) {
    __shared__ float kst[16 * E_DIM];     // recurrent state, f32 (8 KB)
    __shared__ bf16  Abuf[16 * 384];      // A operand staging (12 KB)
    __shared__ int   aIdx[16 * S_DIM];    // a indices, all steps (12.8 KB)
    __shared__ int   sIdx[16 * S_DIM];    // sd indices, all steps (12.8 KB)

    const int b0    = blockIdx.x * 16;
    const int tid   = threadIdx.x;
    const int lane  = tid & 31;
    const int n0    = (tid >> 5) * 16;
    const int colL  = lane & 15;
    const int rbase = (lane >> 4) << 3;

    // ---- loop-invariant preloads -------------------------------------------
    // Weight B-fragments into registers (this wave's 16 output columns).
    v16bf w2f[4], w3f[4], w4f[8], w5f[8], w6f[12];
    #pragma unroll
    for (int i = 0; i < 4; i++) {
        w2f[i] = load_b_frag(W2t, i * 32, n0);
        w3f[i] = load_b_frag(W3t, i * 32, n0);
    }
    #pragma unroll
    for (int i = 0; i < 8; i++) {
        w4f[i] = load_b_frag(W4t, i * 32, n0);
        w5f[i] = load_b_frag(W5t, i * 32, n0);
    }
    #pragma unroll
    for (int i = 0; i < 12; i++)
        w6f[i] = load_b_frag(W6t, i * 32, n0);

    const float vb2 = b2[n0 + colL], vb3 = b3[n0 + colL];
    const float vb4 = b4[n0 + colL], vb5 = b5[n0 + colL];
    const float vb6 = b6[n0 + colL];

    // All per-step gather indices for our 16 rows, loaded once.
    for (int idx = tid; idx < 16 * S_DIM; idx += 256) {
        int r = idx / S_DIM, t = idx % S_DIM;
        aIdx[idx] = aI[(b0 + r) * S_DIM + t];
        sIdx[idx] = sdI[(b0 + r) * S_DIM + t];
    }
    // k0 = broadcast(knowledge)
    for (int idx = tid; idx < 16 * E_DIM; idx += 256)
        kst[idx] = knowledge[idx & 127];
    __syncthreads();

    for (int t = 0; t < S_DIM; ++t) {
        const float* inrow = in_data + ((size_t)t * B_DIM + b0) * E_DIM;
        const float* tgrow = tg_data + ((size_t)t * B_DIM + b0) * E_DIM;
        // Prefetch next step's activation rows (2 x 8 KB) into cache.
        if (t + 1 < S_DIM) {
            __builtin_prefetch(inrow + B_DIM * E_DIM + tid * 8, 0, 0);
            __builtin_prefetch(tgrow + B_DIM * E_DIM + tid * 8, 0, 0);
        }

        // ---- stage qq = k - in_t (cols 0:128), a_t (128:256), sd_t (256:384)
        for (int idx = tid; idx < 16 * E_DIM; idx += 256) {
            int r = idx >> 7, e = idx & 127;
            Abuf[r * 384 + e] = (bf16)(kst[idx] - inrow[idx]);
            int ai = aIdx[r * S_DIM + t];
            int si = sIdx[r * S_DIM + t];
            Abuf[r * 384 + 128 + e] = atb[ai * E_DIM + e];
            Abuf[r * 384 + 256 + e] = sdtb[si * E_DIM + e];
        }
        __syncthreads();

        // ---- GEMM1: qq @ W2^T, qq @ W3^T  (K = 128)
        v8f acc2 = {}, acc3 = {};
        #pragma unroll
        for (int i = 0; i < 4; i++) {
            v16bf a = load_a_frag(Abuf, 384, i * 32);
            acc2 = wmma_bf16(a, w2f[i], acc2);
            acc3 = wmma_bf16(a, w3f[i], acc3);
        }
        __syncthreads();   // all waves done reading qq before overwriting

        // SDFt = sigmoid(.)*tanh(.) -> Abuf cols 0:128 (each wave owns its tile)
        #pragma unroll
        for (int v = 0; v < 8; v++) {
            float g  = sigmoidf_fast(acc2[v] + vb2);
            float th = tanhf(acc3[v] + vb3);
            Abuf[(rbase + v) * 384 + n0 + colL] = (bf16)(g * th);
        }
        __syncthreads();

        // ---- GEMM2: [SDFt, a_t] @ W4^T, W5^T  (K = 256)
        v8f acc4 = {}, acc5 = {};
        #pragma unroll
        for (int i = 0; i < 8; i++) {
            v16bf a = load_a_frag(Abuf, 384, i * 32);
            acc4 = wmma_bf16(a, w4f[i], acc4);
            acc5 = wmma_bf16(a, w5f[i], acc5);
        }
        __syncthreads();   // done reading SDFt before overwriting with k

        float pka[8];
        #pragma unroll
        for (int v = 0; v < 8; v++)
            pka[v] = sigmoidf_fast(acc4[v] + vb4) * tanhf(acc5[v] + vb5);

        // stage k (bf16) into cols 0:128 -> ins = [k, a_t, sd_t]
        for (int idx = tid; idx < 16 * E_DIM; idx += 256)
            Abuf[(idx >> 7) * 384 + (idx & 127)] = (bf16)kst[idx];
        __syncthreads();

        // ---- GEMM3: ins @ W6^T  (K = 384)
        v8f acc6 = {};
        #pragma unroll
        for (int i = 0; i < 12; i++) {
            v16bf a = load_a_frag(Abuf, 384, i * 32);
            acc6 = wmma_bf16(a, w6f[i], acc6);
        }
        // k_new = g*k + (1-g)*PKAt  (each thread owns its 8 state elements)
        #pragma unroll
        for (int v = 0; v < 8; v++) {
            float g    = sigmoidf_fast(acc6[v] + vb6);
            int   sidx = (rbase + v) * E_DIM + n0 + colL;
            float kold = kst[sidx];
            kst[sidx]  = g * kold + (1.0f - g) * pka[v];
        }
        __syncthreads();

        // ---- fused logit: sigmoid(dot(target_data[t], k_new)) -> out[b, t]
        {
            int row = tid >> 4, seg = tid & 15;
            float p = 0.0f;
            #pragma unroll
            for (int j = 0; j < 8; j++) {
                int e = seg * 8 + j;
                p += kst[row * E_DIM + e] * tgrow[row * E_DIM + e];
            }
            #pragma unroll
            for (int off = 8; off >= 1; off >>= 1)
                p += __shfl_xor(p, off, 16);
            if (seg == 0)
                out[(size_t)(b0 + row) * S_DIM + t] = sigmoidf_fast(p);
        }
        // No barrier needed here: the next write to kst is >=5 barriers away.
    }
}

// ---------------------------------------------------------------------------
extern "C" void kernel_launch(void* const* d_in, const int* in_sizes, int n_in,
                              void* d_out, int out_size, void* d_ws, size_t ws_size,
                              hipStream_t stream) {
    (void)in_sizes; (void)n_in; (void)out_size; (void)ws_size;
    // input order: c, sd, a, cshft, sdshft, c_table, sd_table, a_table,
    //              knowledge, W1,b1, W2,b2, W3,b3, W4,b4, W5,b5, W6,b6
    const int*   c       = (const int*)  d_in[0];
    const int*   sd      = (const int*)  d_in[1];
    const int*   a       = (const int*)  d_in[2];
    const int*   cshft   = (const int*)  d_in[3];
    const int*   sdshft  = (const int*)  d_in[4];
    const float* c_table = (const float*)d_in[5];
    const float* sd_table= (const float*)d_in[6];
    const float* a_table = (const float*)d_in[7];
    const float* knowledge = (const float*)d_in[8];
    const float* W1 = (const float*)d_in[9];   const float* b1 = (const float*)d_in[10];
    const float* W2 = (const float*)d_in[11];  const float* b2 = (const float*)d_in[12];
    const float* W3 = (const float*)d_in[13];  const float* b3 = (const float*)d_in[14];
    const float* W4 = (const float*)d_in[15];  const float* b4 = (const float*)d_in[16];
    const float* W5 = (const float*)d_in[17];  const float* b5 = (const float*)d_in[18];
    const float* W6 = (const float*)d_in[19];  const float* b6 = (const float*)d_in[20];
    float* out = (float*)d_out;

    // ---- workspace carve-up -------------------------------------------------
    const size_t SBE = (size_t)S_DIM * B_DIM * E_DIM;     // 13,107,200 f32
    float* in_data = (float*)d_ws;                        // [S][B][E]
    float* tg_data = in_data + SBE;                       // [S][B][E]
    bf16*  W1t  = (bf16*)(tg_data + SBE);                 // [256][128]
    bf16*  W2t  = W1t + 256 * E_DIM;                      // [128][128]
    bf16*  W3t  = W2t + 128 * E_DIM;
    bf16*  W4t  = W3t + 128 * E_DIM;                      // [256][128]
    bf16*  W5t  = W4t + 256 * E_DIM;
    bf16*  W6t  = W5t + 256 * E_DIM;                      // [384][128]
    bf16*  ctb  = W6t + 384 * E_DIM;                      // [1025][128]
    bf16*  sdtb = ctb + 1025 * E_DIM;                     // [102][128]
    bf16*  atb  = sdtb + 102 * E_DIM;                     // [2][128]

    // ---- phase A: bf16 conversions (weights transposed for B-fragments) ----
    auto grd = [](int n) { return dim3((n + 255) / 256); };
    cvt_weightT_kernel<<<grd(256 * E_DIM), 256, 0, stream>>>(W1, W1t, 256);
    cvt_weightT_kernel<<<grd(128 * E_DIM), 256, 0, stream>>>(W2, W2t, 128);
    cvt_weightT_kernel<<<grd(128 * E_DIM), 256, 0, stream>>>(W3, W3t, 128);
    cvt_weightT_kernel<<<grd(256 * E_DIM), 256, 0, stream>>>(W4, W4t, 256);
    cvt_weightT_kernel<<<grd(256 * E_DIM), 256, 0, stream>>>(W5, W5t, 256);
    cvt_weightT_kernel<<<grd(384 * E_DIM), 256, 0, stream>>>(W6, W6t, 384);
    cvt_table_kernel<<<grd(1025 * E_DIM), 256, 0, stream>>>(c_table, ctb, 1025 * E_DIM);
    cvt_table_kernel<<<grd(102 * E_DIM), 256, 0, stream>>>(sd_table, sdtb, 102 * E_DIM);
    cvt_table_kernel<<<grd(2 * E_DIM), 256, 0, stream>>>(a_table, atb, 2 * E_DIM);

    // ---- phase B: W1 projections (gather fused) -----------------------------
    dim3 gB(B_DIM / 16, S_DIM);
    w1_gemm_kernel<<<gB, 256, 0, stream>>>(c, sd, ctb, sdtb, W1t, b1, in_data);
    w1_gemm_kernel<<<gB, 256, 0, stream>>>(cshft, sdshft, ctb, sdtb, W1t, b1, tg_data);

    // ---- phase C: the scan + fused logits -----------------------------------
    dimkt_scan_kernel<<<dim3(B_DIM / 16), 256, 0, stream>>>(
        sd, a, in_data, tg_data, atb, sdtb,
        W2t, W3t, W4t, W5t, W6t,
        b2, b3, b4, b5, b6, knowledge, out);
}
